// BayesConcatSheafLearner_26817775796922
// MI455X (gfx1250) — matrix operand order repaired and checked by reading, not compiled, not run on hardware
//
#include <hip/hip_runtime.h>
#include <stdint.h>

// GEMM view: out[e, 0:256] = A[e, 0:256] @ W  with
//   A[e, 0:128]  = x[row[e]],  A[e, 128:256] = x[col[e]]
//   W[k, n] = w_mean[n][k] (n<128) | w_var[n-128][k] (n>=128)
// out_mean = out[:, :128], out_var = out[:, 128:] (concat-flat output).
// Roofline: 105 GFLOP, ~850 MB HBM (stores dominate; x + W live in 192MB L2).
// f16 inputs + f32 WMMA accumulation puts compute at the ~36us memory floor.

typedef __attribute__((ext_vector_type(16))) _Float16 v16h;
typedef __attribute__((ext_vector_type(8)))  float    v8f;
typedef __attribute__((ext_vector_type(4)))  float    f32x4;
typedef __attribute__((ext_vector_type(4)))  uint32_t u32x4;

#define N_EDGES_C 800000
#define IN_CH_C   128
#define KDIM_C    256
#define ROWPAD_C  264                    // f16 elems per LDS row (pad vs 256)
#define NTILES_C  (N_EDGES_C / 16)       // 50000 M-tiles of 16 edges
#define LDS_HALF  ((16 * ROWPAD_C) / 2)  // u32 words per A-tile buffer

union FragH {
    v16h     v;
    _Float16 h[16];
    uint32_t u[8];
    u32x4    q[2];
};

__device__ __forceinline__ uint32_t pack_f16x2(float lo, float hi) {
    _Float16 l = (_Float16)lo;
    _Float16 h = (_Float16)hi;
    return ((uint32_t)__builtin_bit_cast(uint16_t, h) << 16) |
            (uint32_t)__builtin_bit_cast(uint16_t, l);
}

// LDS-scoped workgroup barrier: flush our DS stores, rendezvous. Unlike
// __syncthreads() this does NOT wait storecnt==0, so the non-temporal HBM
// store stream keeps draining across iterations.
__device__ __forceinline__ void wg_barrier_lds() {
    asm volatile("s_wait_dscnt 0x0\n\t"
                 "s_barrier_signal -1\n\t"
                 "s_barrier_wait -1" ::: "memory");
}

__global__ __launch_bounds__(512, 1)
void sheaf_edge_wmma(const float* __restrict__ x,
                     const int*   __restrict__ ei,
                     const float* __restrict__ w_mean,
                     const float* __restrict__ w_var,
                     float* __restrict__ out)
{
    // double-buffered 16 x 264 f16 A tiles (2 x 8448 B)
    __shared__ uint32_t Atile[2 * LDS_HALF];

    const int tid  = threadIdx.x;
    const int wave = tid >> 5;          // 0..15 -> which 16-wide N tile
    const int lane = tid & 31;
    const int nl   = lane & 15;
    const int half = lane >> 4;

    // ---- B fragments resident in VGPRs for the whole kernel ----
    // B layout (16x16x32 f16): lane%16 = column N = wave*16 + lane%16,
    // elems 0..7 <- k = half*8+0..7, elems 8..15 <- k = 16+half*8+0..7.
    const bool   is_mean = wave < 8;
    const float* wsrc    = is_mean ? w_mean : w_var;
    const int    nb      = (wave & 7) * 16;   // row of w_mean / w_var
    v16h bfrag[8];
    #pragma unroll
    for (int kt = 0; kt < 8; ++kt) {
        FragH f;
        const int k = kt * 32 + half * 8;
        #pragma unroll
        for (int i = 0; i < 16; ++i)
            f.h[i] = (_Float16)wsrc[(size_t)(nb + nl) * KDIM_C + k + (i & 7) + ((i >> 3) << 4)];
        bfrag[kt] = f.v;
    }

    // gather task split: 32 (edge, half) tasks x 16 threads x 8 channels
    const int task  = tid >> 4;         // 0..31
    const int eloc  = task >> 1;        // 0..15 edge within tile
    const int ghalf = task & 1;         // 0: row node, 1: col node
    const int c0    = (tid & 15) * 8;   // first channel handled
    const int lbase = (eloc * ROWPAD_C + ghalf * IN_CH_C + c0) >> 1; // u32 idx

    const size_t var_off = (size_t)N_EDGES_C * IN_CH_C;
    float* obase = out + (is_mean ? (size_t)0 : var_off);
    const int ncol = (wave & 7) * 16 + nl;   // output column within 128

    const int stride = gridDim.x;
    int tile = blockIdx.x;

    // pending gather payload (registers)
    f32x4 ga = {}, gb = {};

    auto gather_load = [&](int t) {
        const int e    = t * 16 + eloc;
        const int node = ei[ghalf * N_EDGES_C + e];
        const float* src = x + (size_t)node * IN_CH_C + c0;
        ga = *(const f32x4*)(src);
        gb = *(const f32x4*)(src + 4);
    };
    auto lds_commit = [&](int buf) {
        u32x4 s;
        s.x = pack_f16x2(ga.x, ga.y);
        s.y = pack_f16x2(ga.z, ga.w);
        s.z = pack_f16x2(gb.x, gb.y);
        s.w = pack_f16x2(gb.z, gb.w);
        *(u32x4*)(Atile + buf * LDS_HALF + lbase) = s;   // 16B-aligned ds b128
    };

    // ---- software pipeline prologue (all guards block-uniform) ----
    if (tile < NTILES_C)          { gather_load(tile); lds_commit(0); }
    if (tile + stride < NTILES_C) { gather_load(tile + stride); }

    int cur = 0;
    for (; tile < NTILES_C; tile += stride) {
        wg_barrier_lds();                        // buf[cur] complete WG-wide

        // stage tile+1 into the spare buffer; prefetch tile+2 from global
        const int t1 = tile + stride;
        if (t1 < NTILES_C)          lds_commit(cur ^ 1);
        if (t1 + stride < NTILES_C) gather_load(t1 + stride);

        // ---- 16(edges) x 16(cols) f32 tile per wave, K = 256 ----
        const uint32_t* abuf = Atile + cur * LDS_HALF;
        v8f acc = {0.f, 0.f, 0.f, 0.f, 0.f, 0.f, 0.f, 0.f};
        #pragma unroll
        for (int kt = 0; kt < 8; ++kt) {
            // A fragment: lane = M (nl); elems 0..7 <- k = kt*32+half*8+0..7,
            // elems 8..15 <- +16..23 (two contiguous 16B runs => ds b128 x2)
            FragH af;
            const int kidx = nl * ROWPAD_C + kt * 32 + half * 8;     // f16 idx
            af.q[0] = *(const u32x4*)(abuf + (kidx >> 1));
            af.q[1] = *(const u32x4*)(abuf + ((kidx + 16) >> 1));
            acc = __builtin_amdgcn_wmma_f32_16x16x32_f16(
                    false, af.v, false, bfrag[kt], (short)0, acc, false, false);
        }

        // ---- C/D layout: VGPR r: lanes 0-15 M=r, lanes 16-31 M=r+8 ----
        {
            float* op = obase + ncol + (size_t)(tile * 16 + half * 8) * IN_CH_C;
            #pragma unroll
            for (int r = 0; r < 8; ++r)
                __builtin_nontemporal_store(acc[r], op + (size_t)r * IN_CH_C);
        }
        cur ^= 1;
    }
}

extern "C" void kernel_launch(void* const* d_in, const int* in_sizes, int n_in,
                              void* d_out, int out_size, void* d_ws, size_t ws_size,
                              hipStream_t stream) {
    const float* x      = (const float*)d_in[0];
    const int*   ei     = (const int*)d_in[1];
    const float* w_mean = (const float*)d_in[2];
    const float* w_var  = (const float*)d_in[3];
    float*       out    = (float*)d_out;

    dim3 grid(2000);   // 25 M-tiles per block; amortizes one-time B load
    dim3 block(512);   // 16 waves: 8 -> mean columns, 8 -> var columns
    hipLaunchKernelGGL(sheaf_edge_wmma, grid, block, 0, stream,
                       x, ei, w_mean, w_var, out);
}
